// SwinTransformerWithKMAttention_42717744726496
// MI455X (gfx1250) — compile-verified, matrix operand order checked by reading, not run on hardware
//
#include <hip/hip_runtime.h>
#include <hip/hip_bf16.h>

typedef _Float16 v16h __attribute__((ext_vector_type(16)));
typedef _Float16 h8   __attribute__((ext_vector_type(8)));
typedef float    v8f  __attribute__((ext_vector_type(8)));

// ---------------- device helpers ----------------
__device__ inline float wred_add(float v){
#pragma unroll
  for (int o = 16; o > 0; o >>= 1) v += __shfl_xor(v, o, 32);
  return v;
}
__device__ inline float wred_max(float v){
#pragma unroll
  for (int o = 16; o > 0; o >>= 1) v = fmaxf(v, __shfl_xor(v, o, 32));
  return v;
}
__device__ inline h8 h8zero(){
  h8 z;
#pragma unroll
  for (int e = 0; e < 8; ++e) z[e] = (_Float16)0.f;
  return z;
}
// Build a 16x32 (or 32x16) f16 WMMA fragment from an LDS row pointer.
// Per CDNA5 ISA 7.12.2: lane group kb = (lane>>4)*8; elements 0..7 = cols kb..kb+7,
// elements 8..15 = cols 16+kb .. 16+kb+7.
__device__ inline v16h ldfrag(const _Float16* rowp, int kb){
  h8 lo = *(const h8*)(rowp + kb);
  h8 hi = *(const h8*)(rowp + 16 + kb);
  v16h f;
#pragma unroll
  for (int e = 0; e < 8; ++e){ f[e] = lo[e]; f[8 + e] = hi[e]; }
  return f;
}
__device__ inline v8f wmma16(v16h a, v16h b, v8f c){
  return __builtin_amdgcn_wmma_f32_16x16x32_f16(false, a, false, b, (short)0, c, false, false);
}

// ---------------- generic WMMA GEMM:  out[M,N] = A[M,K] * W[N,K]^T (+bias)(+gelu)(+res) ----------------
// Block tile 128x64, K-step 64; 8 waves, each wave owns a 2x2 grid of 16x16 tiles
// -> 8 WMMAs per barrier pair per wave.
template<int ACT_GELU, int HAS_RES, int OUT_F16>
__global__ __launch_bounds__(256) void k_gemm(const _Float16* __restrict__ A,
    const _Float16* __restrict__ Wt, const float* __restrict__ bias,
    const float* __restrict__ res, void* __restrict__ outp,
    int M, int N, int K){
  __shared__ _Float16 As[128][72];   // 128 rows x 64 K-cols (+8 pad)
  __shared__ _Float16 Bs[64][72];    // 64  cols x 64 K-cols (+8 pad)
  const int tid = threadIdx.x, lane = tid & 31, wid = tid >> 5;
  const int bm = blockIdx.y * 128, bn = blockIdx.x * 64;
  const int rowq = (wid & 3) * 32;   // two 16-row sub-bands: rowq, rowq+16
  const int colp = (wid >> 2) * 32;  // two 16-col tiles:     colp, colp+16
  v8f acc00 = {0.f,0.f,0.f,0.f,0.f,0.f,0.f,0.f};
  v8f acc01 = acc00, acc10 = acc00, acc11 = acc00;
  const int kb = (lane >> 4) * 8;
  for (int kc = 0; kc < K; kc += 64){
    // stage A: 128x64 halves = 1024 h8-units, 4 per thread
#pragma unroll
    for (int u = 0; u < 4; ++u){
      int unit = tid + u * 256;
      int r = unit >> 3, ccol = (unit & 7) * 8;
      h8 v = h8zero();
      int gm = bm + r;
      if (gm < M && (kc + ccol) < K)
        v = *(const h8*)(A + (size_t)gm * K + kc + ccol);
      *(h8*)&As[r][ccol] = v;
    }
    // stage B: 64x64 halves = 512 h8-units, 2 per thread
#pragma unroll
    for (int u = 0; u < 2; ++u){
      int unit = tid + u * 256;
      int r = unit >> 3, ccol = (unit & 7) * 8;
      h8 v = h8zero();
      int gn = bn + r;
      if (gn < N && (kc + ccol) < K)
        v = *(const h8*)(Wt + (size_t)gn * K + kc + ccol);
      *(h8*)&Bs[r][ccol] = v;
    }
    __syncthreads();
#pragma unroll
    for (int s = 0; s < 2; ++s){   // two 32-K sub-chunks
      v16h a0 = ldfrag(&As[rowq +      (lane & 15)][s * 32], kb);
      v16h a1 = ldfrag(&As[rowq + 16 + (lane & 15)][s * 32], kb);
      v16h b0 = ldfrag(&Bs[colp +      (lane & 15)][s * 32], kb);
      v16h b1 = ldfrag(&Bs[colp + 16 + (lane & 15)][s * 32], kb);
      acc00 = wmma16(a0, b0, acc00);
      acc01 = wmma16(a0, b1, acc01);
      acc10 = wmma16(a1, b0, acc10);
      acc11 = wmma16(a1, b1, acc11);
    }
    __syncthreads();
  }
  const int nl = lane & 15;
#pragma unroll
  for (int i = 0; i < 2; ++i){
#pragma unroll
    for (int j = 0; j < 2; ++j){
      v8f acc = (i == 0) ? (j == 0 ? acc00 : acc01) : (j == 0 ? acc10 : acc11);
      int gn = bn + colp + j * 16 + nl;
      if (gn >= N) continue;
      float bia = bias ? bias[gn] : 0.f;
      int mbase = bm + rowq + i * 16 + (lane >> 4) * 8;
#pragma unroll
      for (int r = 0; r < 8; ++r){
        int gm = mbase + r;
        if (gm >= M) continue;
        float v = acc[r] + bia;
        if (ACT_GELU) v = 0.5f * v * (1.0f + erff(v * 0.70710678118654752f));
        if (HAS_RES)  v += res[(size_t)gm * N + gn];
        if (OUT_F16) ((_Float16*)outp)[(size_t)gm * N + gn] = (_Float16)v;
        else         ((float*)outp)[(size_t)gm * N + gn] = v;
      }
    }
  }
}

// ---------------- per (window, head) attention with q/k LN, top-14 mask, softmax ----------------
__global__ __launch_bounds__(128) void k_attn(const _Float16* __restrict__ qkv,
    _Float16* __restrict__ outb,
    const float* __restrict__ qg, const float* __restrict__ qb,
    const float* __restrict__ kg, const float* __restrict__ kb2,
    int C, int nh){
  __shared__ _Float16 qs[64][72];  // q, later attention probs (A operand)
  __shared__ _Float16 ks[64][72];  // k
  __shared__ _Float16 vt[48][72];  // v transposed: vt[d][token]
  __shared__ float    S[64][68];   // scores
  const int tid = threadIdx.x, lane = tid & 31, wid = tid >> 5;
  const int win = blockIdx.x / nh, head = blockIdx.x % nh;
  {
    unsigned* z1 = (unsigned*)&qs[0][0];
    for (int i = tid; i < 64 * 36; i += 128) z1[i] = 0u;
    unsigned* z2 = (unsigned*)&ks[0][0];
    for (int i = tid; i < 64 * 36; i += 128) z2[i] = 0u;
    unsigned* z3 = (unsigned*)&vt[0][0];
    for (int i = tid; i < 48 * 36; i += 128) z3[i] = 0u;
  }
  __syncthreads();
  const float scale = rsqrtf(48.f);   // hd^-0.5 folded into q
  const int d0 = lane, d1 = lane + 32;
  for (int t = wid; t < 49; t += 4){
    size_t rb = ((size_t)win * 49 + t) * (size_t)(3 * C) + (size_t)head * 48;
    float q0 = (float)qkv[rb + d0];
    float q1 = (lane < 16) ? (float)qkv[rb + d1] : 0.f;
    float s  = wred_add(q0 + q1);
    float ss = wred_add(q0 * q0 + q1 * q1);
    float mean = s * (1.f / 48.f);
    float var  = ss * (1.f / 48.f) - mean * mean;
    float rstd = rsqrtf(var + 1e-5f);
    qs[t][d0] = (_Float16)(((q0 - mean) * rstd * qg[d0] + qb[d0]) * scale);
    if (lane < 16) qs[t][d1] = (_Float16)(((q1 - mean) * rstd * qg[d1] + qb[d1]) * scale);
    float k0 = (float)qkv[rb + C + d0];
    float k1 = (lane < 16) ? (float)qkv[rb + C + d1] : 0.f;
    s  = wred_add(k0 + k1);
    ss = wred_add(k0 * k0 + k1 * k1);
    mean = s * (1.f / 48.f);
    var  = ss * (1.f / 48.f) - mean * mean;
    rstd = rsqrtf(var + 1e-5f);
    ks[t][d0] = (_Float16)((k0 - mean) * rstd * kg[d0] + kb2[d0]);
    if (lane < 16) ks[t][d1] = (_Float16)((k1 - mean) * rstd * kg[d1] + kb2[d1]);
    vt[d0][t] = qkv[rb + 2 * C + d0];
    if (lane < 16) vt[d1][t] = qkv[rb + 2 * C + d1];
  }
  __syncthreads();
  const int kbh = (lane >> 4) * 8;
  const int mr  = wid * 16 + (lane & 15);
  // S = q * k^T (64x64, K=64)
  for (int nt = 0; nt < 4; ++nt){
    v8f acc = {0.f,0.f,0.f,0.f,0.f,0.f,0.f,0.f};
    for (int kc = 0; kc < 64; kc += 32){
      v16h a = ldfrag(&qs[mr][kc], kbh);
      v16h b = ldfrag(&ks[nt * 16 + (lane & 15)][kc], kbh);
      acc = wmma16(a, b, acc);
    }
    int nl = lane & 15, mb = wid * 16 + (lane >> 4) * 8;
#pragma unroll
    for (int r = 0; r < 8; ++r) S[mb + r][nt * 16 + nl] = acc[r];
  }
  __syncthreads();
  const float NINF = -__builtin_inff();
  // top-14 threshold + softmax per row; write probs into qs (rows>=49 stay zero)
  for (int row = wid; row < 49; row += 4){
    int c0 = lane * 2, c1 = c0 + 1;
    float x0 = (c0 < 49) ? S[row][c0] : NINF;
    float x1 = (c1 < 49) ? S[row][c1] : NINF;
    float thr = __builtin_inff(), rmax = 0.f;
#pragma unroll 1
    for (int it = 0; it < 14; ++it){   // kv = int(49*0.3) = 14
      float m = fmaxf(x0 < thr ? x0 : NINF, x1 < thr ? x1 : NINF);
      m = wred_max(m);
      if (it == 0) rmax = m;
      thr = m;
    }
    float p0 = (c0 < 49 && x0 >= thr) ? __expf(x0 - rmax) : 0.f;
    float p1 = (c1 < 49 && x1 >= thr) ? __expf(x1 - rmax) : 0.f;
    float inv = 1.f / wred_add(p0 + p1);
    qs[row][c0] = (_Float16)(p0 * inv);
    qs[row][c1] = (_Float16)(p1 * inv);
  }
  __syncthreads();
  // O = P * V  (64x48, K=64)
  for (int nt = 0; nt < 3; ++nt){
    v8f acc = {0.f,0.f,0.f,0.f,0.f,0.f,0.f,0.f};
    for (int kc = 0; kc < 64; kc += 32){
      v16h a = ldfrag(&qs[mr][kc], kbh);
      v16h b = ldfrag(&vt[nt * 16 + (lane & 15)][kc], kbh);
      acc = wmma16(a, b, acc);
    }
    int nl = lane & 15, mb = wid * 16 + (lane >> 4) * 8;
#pragma unroll
    for (int r = 0; r < 8; ++r){
      int m = mb + r;
      if (m < 49)
        outb[((size_t)win * 49 + m) * C + head * 48 + nt * 16 + nl] = (_Float16)acc[r];
    }
  }
}

// ---------------- LN + (optional cyclic shift via wrap-pad) + window partition ----------------
__global__ __launch_bounds__(128) void k_part_ln(const float* __restrict__ Y,
    _Float16* __restrict__ xw, const float* __restrict__ g, const float* __restrict__ b_,
    int H, int C, int pad, int nwh){
  const int tid = threadIdx.x, lane = tid & 31, wid = tid >> 5;
  int t = blockIdx.x % 49;
  int win = blockIdx.x / 49;
  int wx = win % nwh; int r2 = win / nwh; int wy = r2 % nwh; int b = r2 / nwh;
  int hp = wy * 7 + t / 7, wp = wx * 7 + t % 7;
  int Hs = H + pad;
  _Float16* orow = xw + (size_t)blockIdx.x * C;
  if (hp >= Hs || wp >= Hs){                         // zero-pad region of _win_part
    for (int c = tid; c < C; c += 128) orow[c] = (_Float16)0.f;
    return;
  }
  int h = (hp + H - pad) % H;                        // wrap pad (pad rows/cols before)
  int w = (wp + H - pad) % H;
  const float* irow = Y + ((size_t)(b * H + h) * H + w) * (size_t)C;
  float s = 0.f, ss = 0.f;
  for (int c = tid; c < C; c += 128){ float v = irow[c]; s += v; ss += v * v; }
  s = wred_add(s); ss = wred_add(ss);
  __shared__ float sh[8];
  if (lane == 0){ sh[wid] = s; sh[4 + wid] = ss; }
  __syncthreads();
  float S_ = sh[0] + sh[1] + sh[2] + sh[3];
  float SS = sh[4] + sh[5] + sh[6] + sh[7];
  float mean = S_ / C, var = SS / C - mean * mean, rstd = rsqrtf(var + 1e-5f);
  for (int c = tid; c < C; c += 128)
    orow[c] = (_Float16)((irow[c] - mean) * rstd * g[c] + b_[c]);
}

// ---------------- plain row LayerNorm -> f16 ----------------
__global__ __launch_bounds__(128) void k_ln_rows(const float* __restrict__ in,
    _Float16* __restrict__ out, const float* __restrict__ g, const float* __restrict__ b_, int C){
  const int tid = threadIdx.x, lane = tid & 31, wid = tid >> 5;
  const float* ir = in + (size_t)blockIdx.x * C;
  _Float16* orr = out + (size_t)blockIdx.x * C;
  float s = 0.f, ss = 0.f;
  for (int c = tid; c < C; c += 128){ float v = ir[c]; s += v; ss += v * v; }
  s = wred_add(s); ss = wred_add(ss);
  __shared__ float sh[8];
  if (lane == 0){ sh[wid] = s; sh[4 + wid] = ss; }
  __syncthreads();
  float S_ = sh[0] + sh[1] + sh[2] + sh[3];
  float SS = sh[4] + sh[5] + sh[6] + sh[7];
  float mean = S_ / C, var = SS / C - mean * mean, rstd = rsqrtf(var + 1e-5f);
  for (int c = tid; c < C; c += 128)
    orr[c] = (_Float16)((ir[c] - mean) * rstd * g[c] + b_[c]);
}

// ---------------- window reverse + crop + residual add ----------------
__global__ void k_rev_add(const float* __restrict__ yin, const float* __restrict__ D,
    float* __restrict__ yout, int H, int C, int pad, int nwh, int total){
  int idx = blockIdx.x * 256 + threadIdx.x;
  if (idx >= total) return;
  int c = idx % C;
  int l = (idx / C) % (H * H);
  int b = idx / (C * H * H);
  int h = l / H, w = l % H;
  int hp = h + pad, wp = w + pad;
  int win = (b * nwh + hp / 7) * nwh + wp / 7;
  int pos = (hp % 7) * 7 + (wp % 7);
  yout[idx] = yin[idx] + D[((size_t)win * 49 + pos) * C + c];
}

// ---------------- patch embed: conv4x4/4 + bias + patch-norm LN ----------------
__global__ __launch_bounds__(96) void k_patch(const float* __restrict__ x,
    const float* __restrict__ cw, const float* __restrict__ cb,
    const float* __restrict__ pg, const float* __restrict__ pb,
    float* __restrict__ Y){
  __shared__ float patch[48];
  __shared__ float sh[6];
  int tid = threadIdx.x, lane = tid & 31, wid = tid >> 5;
  int bid = blockIdx.x;
  int w = bid % 56; int r2 = bid / 56; int h = r2 % 56; int b = r2 / 56;
  if (tid < 48){
    int c = tid >> 4; int dy = (tid >> 2) & 3; int dx = tid & 3;
    patch[tid] = x[(((size_t)b * 3 + c) * 224 + (h * 4 + dy)) * 224 + (w * 4 + dx)];
  }
  __syncthreads();
  float acc = cb[tid];
#pragma unroll
  for (int k = 0; k < 48; ++k) acc += patch[k] * cw[tid * 48 + k];
  float s = wred_add(acc), ss = wred_add(acc * acc);
  if (lane == 0){ sh[wid] = s; sh[3 + wid] = ss; }
  __syncthreads();
  float S_ = sh[0] + sh[1] + sh[2], SS = sh[3] + sh[4] + sh[5];
  float mean = S_ * (1.f / 96.f), var = SS * (1.f / 96.f) - mean * mean;
  float rstd = rsqrtf(var + 1e-5f);
  Y[((size_t)b * 3136 + h * 56 + w) * 96 + tid] = (acc - mean) * rstd * pg[tid] + pb[tid];
}

// ---------------- patch merging: 2x2 gather + LN -> f16 [B*L/4, 4C] ----------------
__global__ __launch_bounds__(256) void k_merge(const float* __restrict__ Y,
    _Float16* __restrict__ out, const float* __restrict__ g, const float* __restrict__ b_,
    int H, int C){
  __shared__ float buf[1536];
  __shared__ float sh[16];
  int C4 = 4 * C; int W2 = H / 2;
  int row = blockIdx.x;
  int j = row % W2; int t2 = row / W2; int i = t2 % W2; int b = t2 / W2;
  int tid = threadIdx.x, lane = tid & 31, wid = tid >> 5;
  float s = 0.f, ss = 0.f;
  for (int oc = tid; oc < C4; oc += 256){
    int chunk = oc / C, c = oc - chunk * C;
    int hh = 2 * i + (chunk & 1), ww = 2 * j + (chunk >> 1);
    float v = Y[((size_t)(b * H + hh) * H + ww) * C + c];
    buf[oc] = v; s += v; ss += v * v;
  }
  s = wred_add(s); ss = wred_add(ss);
  if (lane == 0){ sh[wid] = s; sh[8 + wid] = ss; }
  __syncthreads();
  float S_ = 0.f, SS = 0.f;
  for (int q = 0; q < 8; ++q){ S_ += sh[q]; SS += sh[8 + q]; }
  float mean = S_ / C4, var = SS / C4 - mean * mean, rstd = rsqrtf(var + 1e-5f);
  for (int oc = tid; oc < C4; oc += 256)
    out[(size_t)row * C4 + oc] = (_Float16)((buf[oc] - mean) * rstd * g[oc] + b_[oc]);
}

// ---------------- final LN + mean over 49 tokens ----------------
__global__ __launch_bounds__(256) void k_final(const float* __restrict__ Y,
    const float* __restrict__ g, const float* __restrict__ b_, float* __restrict__ out){
  __shared__ float acc[768];
  __shared__ float sh[16];
  int b = blockIdx.x;
  int tid = threadIdx.x, lane = tid & 31, wid = tid >> 5;
  for (int c = tid; c < 768; c += 256) acc[c] = 0.f;
  __syncthreads();
  for (int t = 0; t < 49; ++t){
    const float* r = Y + ((size_t)b * 49 + t) * 768;
    float s = 0.f, ss = 0.f;
    for (int c = tid; c < 768; c += 256){ float v = r[c]; s += v; ss += v * v; }
    s = wred_add(s); ss = wred_add(ss);
    if (lane == 0){ sh[wid] = s; sh[8 + wid] = ss; }
    __syncthreads();
    float S_ = 0.f, SS = 0.f;
    for (int q = 0; q < 8; ++q){ S_ += sh[q]; SS += sh[8 + q]; }
    float mean = S_ * (1.f / 768.f), var = SS * (1.f / 768.f) - mean * mean;
    float rstd = rsqrtf(var + 1e-5f);
    for (int c = tid; c < 768; c += 256)
      acc[c] += (r[c] - mean) * rstd * g[c] + b_[c];
    __syncthreads();
  }
  for (int c = tid; c < 768; c += 256) out[(size_t)b * 768 + c] = acc[c] * (1.f / 49.f);
}

__global__ void k_cast(const float* __restrict__ s, _Float16* __restrict__ d, int n){
  int i = blockIdx.x * 256 + threadIdx.x;
  if (i < n) d[i] = (_Float16)s[i];
}

// =============================== host ===============================
struct LayerPtrs {
  const float *n1_g,*n1_b,*qkv_w,*qkv_b,*qn_g,*qn_b,*kn_g,*kn_b,
              *pr_w,*pr_b,*n2_g,*n2_b,*f1_w,*f1_b,*f2_w,*f2_b;
};

extern "C" void kernel_launch(void* const* d_in, const int* in_sizes, int n_in,
                              void* d_out, int out_size, void* d_ws, size_t ws_size,
                              hipStream_t stream){
  (void)out_size;
  if (n_in < 112) return;
  const float *X, *conv_w, *conv_b, *pn_g, *pn_b, *fn_g, *fn_b;
  LayerPtrs LY[6];
  const float* DSP[3][3];   // [i]: {dn_g, dn_b, rd_w}
  int p = 0;
  auto nxt = [&](){ return (const float*)d_in[p++]; };
  bool xfirst = (in_sizes[0] == 64 * 3 * 224 * 224);   // dict-insertion-order flatten
  if (xfirst){
    X = nxt(); conv_w = nxt(); conv_b = nxt(); pn_g = nxt(); pn_b = nxt();
    for (int l = 0; l < 6; ++l){ LayerPtrs& q = LY[l];
      q.n1_g=nxt(); q.n1_b=nxt(); q.qkv_w=nxt(); q.qkv_b=nxt(); q.qn_g=nxt(); q.qn_b=nxt();
      q.kn_g=nxt(); q.kn_b=nxt(); q.pr_w=nxt(); q.pr_b=nxt(); q.n2_g=nxt(); q.n2_b=nxt();
      q.f1_w=nxt(); q.f1_b=nxt(); q.f2_w=nxt(); q.f2_b=nxt(); }
    for (int d = 0; d < 3; ++d){ DSP[d][0]=nxt(); DSP[d][1]=nxt(); DSP[d][2]=nxt(); }
    fn_g = nxt(); fn_b = nxt();
  } else {                                              // jax sorted-key flatten
    conv_b = nxt(); conv_w = nxt();
    for (int d = 0; d < 3; ++d){ DSP[d][1]=nxt(); DSP[d][0]=nxt(); DSP[d][2]=nxt(); }
    fn_b = nxt(); fn_g = nxt();
    for (int l = 0; l < 6; ++l){ LayerPtrs& q = LY[l];
      q.f1_b=nxt(); q.f1_w=nxt(); q.f2_b=nxt(); q.f2_w=nxt(); q.kn_b=nxt(); q.kn_g=nxt();
      q.n1_b=nxt(); q.n1_g=nxt(); q.n2_b=nxt(); q.n2_g=nxt(); q.pr_b=nxt(); q.pr_w=nxt();
      q.qkv_b=nxt(); q.qkv_w=nxt(); q.qn_b=nxt(); q.qn_g=nxt(); }
    pn_b = nxt(); pn_g = nxt(); X = nxt();
  }

  // ---- workspace arena ----
  char* base = (char*)d_ws; size_t off = 0;
  auto alloc = [&](size_t bytes)->char*{
    char* r = base + off; off = (off + bytes + 255) & ~(size_t)255; return r;
  };
  int dims[4]   = {96, 192, 384, 768};
  int heads_[4] = {2, 4, 8, 16};
  int depths_[4]= {1, 1, 3, 1};
  _Float16 *qkvW[6], *prW[6], *f1W[6], *f2W[6], *rdW[3];
  {
    int li = 0;
    for (int i = 0; i < 4; ++i){
      int C = dims[i], hid = 3 * C;
      for (int j = 0; j < depths_[i]; ++j, ++li){
        qkvW[li] = (_Float16*)alloc((size_t)3 * C * C * 2);
        prW[li]  = (_Float16*)alloc((size_t)C * C * 2);
        f1W[li]  = (_Float16*)alloc((size_t)hid * C * 2);
        f2W[li]  = (_Float16*)alloc((size_t)C * hid * 2);
      }
      if (i < 3) rdW[i] = (_Float16*)alloc((size_t)8 * C * C * 2);
    }
  }
  float*    Y0   = (float*)alloc((size_t)64 * 3136 * 96 * 4);
  float*    Y1   = (float*)alloc((size_t)64 * 3136 * 96 * 4);
  _Float16* bufA = (_Float16*)alloc((size_t)4096 * 49 * 96 * 2);   // xw / attn-out / LN2 / merge
  _Float16* bufB = (_Float16*)alloc((size_t)4096 * 49 * 288 * 2);  // qkv / proj-f32 / mlp-hidden
  if (off > ws_size) return;   // insufficient scratch

  auto cast = [&](const float* s, _Float16* d, size_t n){
    k_cast<<<(unsigned)((n + 255) / 256), 256, 0, stream>>>(s, d, (int)n);
  };
  {
    int li = 0;
    for (int i = 0; i < 4; ++i){
      int C = dims[i], hid = 3 * C;
      for (int j = 0; j < depths_[i]; ++j, ++li){
        cast(LY[li].qkv_w, qkvW[li], (size_t)3 * C * C);
        cast(LY[li].pr_w,  prW[li],  (size_t)C * C);
        cast(LY[li].f1_w,  f1W[li],  (size_t)hid * C);
        cast(LY[li].f2_w,  f2W[li],  (size_t)C * hid);
      }
      if (i < 3) cast(DSP[i][2], rdW[i], (size_t)8 * C * C);
    }
  }

  // ---- forward ----
  k_patch<<<64 * 3136, 96, 0, stream>>>(X, conv_w, conv_b, pn_g, pn_b, Y0);
  float* ycur = Y0; float* yalt = Y1;
  int H = 56, li = 0;
  for (int i = 0; i < 4; ++i){
    int C = dims[i], nh = heads_[i], hid = 3 * C;
    int L = H * H, Mtok = 64 * L;
    for (int j = 0; j < depths_[i]; ++j, ++li){
      int shift = ((i + j) & 1) ? 3 : 0;
      int pad = shift ? (7 - shift) : 0;
      int nwh = (H + pad + 6) / 7, NWIN = 64 * nwh * nwh, Mw = NWIN * 49;
      const LayerPtrs& q = LY[li];
      k_part_ln<<<Mw, 128, 0, stream>>>(ycur, bufA, q.n1_g, q.n1_b, H, C, pad, nwh);
      { dim3 g((3 * C + 63) / 64, (Mw + 127) / 128);
        k_gemm<0,0,1><<<g, 256, 0, stream>>>(bufA, qkvW[li], q.qkv_b, nullptr, bufB, Mw, 3 * C, C); }
      k_attn<<<NWIN * nh, 128, 0, stream>>>(bufB, bufA, q.qn_g, q.qn_b, q.kn_g, q.kn_b, C, nh);
      { dim3 g((C + 63) / 64, (Mw + 127) / 128);
        k_gemm<0,0,0><<<g, 256, 0, stream>>>(bufA, prW[li], q.pr_b, nullptr, (void*)bufB, Mw, C, C); }
      { int total = Mtok * C;
        k_rev_add<<<(total + 255) / 256, 256, 0, stream>>>(ycur, (const float*)bufB, yalt, H, C, pad, nwh, total); }
      k_ln_rows<<<Mtok, 128, 0, stream>>>(yalt, bufA, q.n2_g, q.n2_b, C);
      { dim3 g((hid + 63) / 64, (Mtok + 127) / 128);
        k_gemm<1,0,1><<<g, 256, 0, stream>>>(bufA, f1W[li], q.f1_b, nullptr, bufB, Mtok, hid, C); }
      { dim3 g((C + 63) / 64, (Mtok + 127) / 128);
        k_gemm<0,1,0><<<g, 256, 0, stream>>>(bufB, f2W[li], q.f2_b, yalt, (void*)ycur, Mtok, C, hid); }
    }
    if (i < 3){
      int M2 = Mtok / 4;
      k_merge<<<M2, 256, 0, stream>>>(ycur, bufA, DSP[i][0], DSP[i][1], H, C);
      { dim3 g((2 * C + 63) / 64, (M2 + 127) / 128);
        k_gemm<0,0,0><<<g, 256, 0, stream>>>(bufA, rdW[i], nullptr, nullptr, (void*)yalt, M2, 2 * C, 4 * C); }
      float* t = ycur; ycur = yalt; yalt = t;
      H >>= 1;
    }
  }
  k_final<<<64, 256, 0, stream>>>(ycur, fn_g, fn_b, (float*)d_out);
}
